// GPT2_48301202211024
// MI455X (gfx1250) — matrix operand order repaired and checked by reading, not compiled, not run on hardware
//
#include <hip/hip_runtime.h>
#include <math.h>

#define T_SEQ  1024
#define DMODEL 768
#define NHEAD  12
#define HSZ    64
#define NLAYER 12
#define DFF    3072
#define VOCAB  50257

typedef __attribute__((ext_vector_type(16))) _Float16 v16h;
typedef __attribute__((ext_vector_type(8)))  float    v8f;
typedef __attribute__((ext_vector_type(4)))  unsigned int v4u;
typedef __attribute__((ext_vector_type(8)))  int      v8i;
typedef __attribute__((ext_vector_type(4)))  int      v4i;

union Frag16 { v16h v; uint4 q[2]; };
union H8     { uint4 q; _Float16 h[8]; };

enum { EPI_F32 = 0, EPI_F16_BIAS = 1, EPI_F16_GELU = 2, EPI_RESID = 3 };
enum { B_FP32_KN = 0, B_F16_NK = 1, B_F16_KN = 2 };

#define BM   128
#define BN   64
#define BK   32
#define LDSS 40   // padded LDS row stride in halves (64B data + 16B pad)

// ---- Tensor Data Mover availability (device pass only; safe fallback) ----
#if defined(__AMDGCN__) && defined(__has_builtin)
#  if __has_builtin(__builtin_amdgcn_tensor_load_to_lds) && \
      __has_builtin(__builtin_amdgcn_s_wait_tensorcnt)
#    define HAVE_TDM 1
#  endif
#endif
#ifndef HAVE_TDM
#  define HAVE_TDM 0
#endif

#if HAVE_TDM
// Issue one TDM 2D tile load: tileRows x tileElems (f16) from global (row
// stride strideElems) into LDS at ldsOff, hardware-padding each 64B row by
// 16B so the LDS image has the LDSS=40-half stride the fragment gathers use.
__device__ __forceinline__ void tdm_load_2d(unsigned ldsOff, const void* gsrc,
                                            int tileElems, int tileRows,
                                            long strideElems)
{
  const unsigned long ga = (unsigned long)gsrc;
  const unsigned TDIM = 1u << 20;   // generous tensor bounds (tiles never OOB)
  v4u g0;
  g0.x = 1u;                                                // count=1, user D#
  g0.y = ldsOff;                                            // lds_addr (bytes)
  g0.z = (unsigned)ga;                                      // global_addr[31:0]
  g0.w = ((unsigned)(ga >> 32) & 0x01ffffffu) | (2u << 30); // addr[56:32]|type=2
  v8i g1;
  g1[0] = (1 << 16)        // data_size = 2 bytes
        | (1 << 20)        // pad_enable
        | (3 << 22)        // pad_interval: 16 DWORDs (64B) between pads
        | (3 << 25);       // pad_amount:   4 DWORDs (16B)
  g1[1] = (int)((TDIM & 0xffffu) << 16);                      // tensor_dim0 lo16
  g1[2] = (int)((TDIM >> 16) | ((TDIM & 0xffffu) << 16));     // dim0 hi | dim1 lo
  g1[3] = (int)((TDIM >> 16) | ((unsigned)tileElems << 16));  // dim1 hi | tile0
  g1[4] = tileRows & 0xffff;                                  // tile1 | tile2=0
  g1[5] = (int)(unsigned)strideElems;                         // dim0_stride lo32
  g1[6] = (int)((unsigned)((unsigned long)strideElems >> 32) & 0xffffu);
  g1[7] = 0;                                                  // dim1_stride
  v4i gz = {0, 0, 0, 0};
#if __clang_major__ >= 23
  v8i gz8 = {0, 0, 0, 0, 0, 0, 0, 0};
  __builtin_amdgcn_tensor_load_to_lds(g0, g1, gz, gz, gz8, 0);
#else
  __builtin_amdgcn_tensor_load_to_lds(g0, g1, gz, gz, 0);
#endif
}
#endif

// ---------------------------------------------------------------------------
// Tiled WMMA GEMM: C[M,N] (+epilogue) = A[M,K](f16) * B[K,N]
//   256 threads = 8 waves; wave grid 4x2; each wave owns 32x32 (2x2 WMMA).
//   A (and B in NK mode) staged by the Tensor Data Mover when available.
// ---------------------------------------------------------------------------
template <int EPI, int BMODE, int RAGGED = 0>
__global__ __launch_bounds__(256)
void gemm_wmma(const _Float16* __restrict__ A, long aBatch, int lda,
               const void* __restrict__ Bp, long bBatch, int ldb,
               const float* __restrict__ bias, long biasBatch,
               float* __restrict__ Cf, long cfBatch, int ldc,
               _Float16* __restrict__ Ch, long chBatch, int ldch,
               int N, int K, int causal)
{
  const int bm = blockIdx.x, bn = blockIdx.y, bz = blockIdx.z;
  if (causal && bn * BN > bm * BM + (BM - 1)) return;   // fully-masked block

  A += (long)bz * aBatch;
  const float*    Bf = (const float*)Bp    + ((BMODE == B_FP32_KN) ? (long)bz * bBatch : 0);
  const _Float16* Bh = (const _Float16*)Bp + ((BMODE != B_FP32_KN) ? (long)bz * bBatch : 0);
  const float* biasP = bias ? bias + (long)bz * biasBatch : nullptr;
  float*    CfP = Cf ? Cf + (long)bz * cfBatch : nullptr;
  _Float16* ChP = Ch ? Ch + (long)bz * chBatch : nullptr;

  __shared__ _Float16 As[BM * LDSS];
  __shared__ _Float16 Bs[BN * LDSS];

  const int tid  = threadIdx.x;
  const int lane = tid & 31;
  const int wave = tid >> 5;
  const int wm   = wave & 3;    // wave row: 32 rows each
  const int wn   = wave >> 2;   // wave col: 32 cols each
  const int l16  = lane & 15;
  const int lhi  = lane >> 4;

#if HAVE_TDM
  const unsigned asOff = (unsigned)(unsigned long)(const void*)As;
  const unsigned bsOff = (unsigned)(unsigned long)(const void*)Bs;
#endif

  v8f acc[2][2] = {};

  for (int k0 = 0; k0 < K; k0 += BK) {
    // ---- stage A tile (BM x BK halves) ----
#if HAVE_TDM
    if (wave == 0) {
      tdm_load_2d(asOff, A + (long)(bm * BM) * lda + k0, BK, BM, lda);
      if (BMODE == B_F16_NK)
        tdm_load_2d(bsOff, Bh + (long)(bn * BN) * ldb + k0, BK, BN, ldb);
      __builtin_amdgcn_s_wait_tensorcnt(0);
    }
#else
    {
      const int row = tid >> 1;
      const int hc  = (tid & 1) * 16;
      const _Float16* src = A + (long)(bm * BM + row) * lda + k0 + hc;
      *(uint4*)&As[row * LDSS + hc]     = *(const uint4*)(src);
      *(uint4*)&As[row * LDSS + hc + 8] = *(const uint4*)(src + 8);
    }
#endif
    // ---- stage B tile into Bs[n][k] ----
    if (BMODE == B_FP32_KN) {          // fp32 weights [K][N]: convert+transpose
      const int kr = tid >> 3;         // 0..31
      const int cg = (tid & 7) * 8;    // 0..56
      const float* src = Bf + (long)(k0 + kr) * ldb + bn * BN + cg;
      if (k0 + BK < K)
        __builtin_prefetch(src + (long)BK * ldb, 0, 1);   // global_prefetch_b8
      if (!RAGGED) {                   // aligned: 2 x b128 weight loads
        const float4 f0 = *(const float4*)(src);
        const float4 f1 = *(const float4*)(src + 4);
        _Float16* d = &Bs[cg * LDSS + kr];
        d[0 * LDSS] = (_Float16)f0.x; d[1 * LDSS] = (_Float16)f0.y;
        d[2 * LDSS] = (_Float16)f0.z; d[3 * LDSS] = (_Float16)f0.w;
        d[4 * LDSS] = (_Float16)f1.x; d[5 * LDSS] = (_Float16)f1.y;
        d[6 * LDSS] = (_Float16)f1.z; d[7 * LDSS] = (_Float16)f1.w;
      } else {                         // lm_head tail: guarded scalar loads
        #pragma unroll
        for (int i = 0; i < 8; ++i) {
          const int col = bn * BN + cg + i;
          Bs[(cg + i) * LDSS + kr] = (_Float16)((col < N) ? src[i] : 0.0f);
        }
      }
    } else if (BMODE == B_F16_NK) {    // f16 stored [N][K] (k^T): direct copy
#if !HAVE_TDM
      if (tid < 128) {
        const int row = tid >> 1;
        const int hc  = (tid & 1) * 16;
        const _Float16* src = Bh + (long)(bn * BN + row) * ldb + k0 + hc;
        *(uint4*)&Bs[row * LDSS + hc]     = *(const uint4*)(src);
        *(uint4*)&Bs[row * LDSS + hc + 8] = *(const uint4*)(src + 8);
      }
#endif
    } else {                           // f16 stored [K][N] (v): transpose
      const int kr = tid >> 3;
      const int cg = (tid & 7) * 8;
      H8 u;
      u.q = *(const uint4*)(Bh + (long)(k0 + kr) * ldb + bn * BN + cg);
      #pragma unroll
      for (int i = 0; i < 8; ++i) Bs[(cg + i) * LDSS + kr] = u.h[i];
    }
    __syncthreads();

    // ---- gather fragments per CDNA5 WMMA VGPR layouts ----
    Frag16 af[2], bf[2];
    #pragma unroll
    for (int mi = 0; mi < 2; ++mi) {
      // A: M=lane%16; lanes<16 hold K 0-7,16-23; lanes>=16 hold K 8-15,24-31
      const _Float16* p = &As[(wm * 32 + mi * 16 + l16) * LDSS + lhi * 8];
      af[mi].q[0] = *(const uint4*)(p);
      af[mi].q[1] = *(const uint4*)(p + 16);
    }
    #pragma unroll
    for (int ni = 0; ni < 2; ++ni) {
      // B: N=lane%16; lanes<16 hold K 0-7,16-23; lanes>=16 hold K 8-15,24-31
      const _Float16* p = &Bs[(wn * 32 + ni * 16 + l16) * LDSS + lhi * 8];
      bf[ni].q[0] = *(const uint4*)(p);
      bf[ni].q[1] = *(const uint4*)(p + 16);
    }
    #pragma unroll
    for (int mi = 0; mi < 2; ++mi)
      #pragma unroll
      for (int ni = 0; ni < 2; ++ni)
        acc[mi][ni] = __builtin_amdgcn_wmma_f32_16x16x32_f16(
            false, af[mi].v, false, bf[ni].v, (short)0, acc[mi][ni], false, false);
    __syncthreads();
  }

  // ---- epilogue : D layout M = r + 8*(lane>=16), N = lane%16 ----
  #pragma unroll
  for (int ni = 0; ni < 2; ++ni) {
    const int col = bn * BN + wn * 32 + ni * 16 + l16;
    if (col >= N) continue;
    const float bv = biasP ? biasP[col] : 0.0f;
    #pragma unroll
    for (int mi = 0; mi < 2; ++mi) {
      #pragma unroll
      for (int r = 0; r < 8; ++r) {
        const int row = bm * BM + wm * 32 + mi * 16 + lhi * 8 + r;
        const float v = acc[mi][ni][r] + bv;
        if (EPI == EPI_F32) {
          CfP[(long)row * ldc + col] = v;
        } else if (EPI == EPI_F16_BIAS) {
          ChP[(long)row * ldch + col] = (_Float16)v;
        } else if (EPI == EPI_F16_GELU) {
          ChP[(long)row * ldch + col] =
              (_Float16)(0.5f * v * (1.0f + erff(v * 0.7071067811865475f)));
        } else {  // EPI_RESID
          float* p = &CfP[(long)row * ldc + col];
          *p = *p + v;
        }
      }
    }
  }
}

// ---------------------------------------------------------------------------
// Row LayerNorm (fp32 in -> f16 out, the WMMA A operand feed)
// ---------------------------------------------------------------------------
__global__ __launch_bounds__(256)
void layernorm_f16(const float* __restrict__ x, const float* __restrict__ sc,
                   const float* __restrict__ bi, _Float16* __restrict__ out)
{
  const int t = blockIdx.x;
  const float* row = x + (long)t * DMODEL;
  float s = 0.f, s2 = 0.f;
  for (int d = threadIdx.x; d < DMODEL; d += 256) {
    const float v = row[d];
    s += v; s2 += v * v;
  }
  #pragma unroll
  for (int o = 16; o > 0; o >>= 1) {
    s  += __shfl_xor(s,  o, 32);
    s2 += __shfl_xor(s2, o, 32);
  }
  __shared__ float ps[8], ps2[8];
  if ((threadIdx.x & 31) == 0) { ps[threadIdx.x >> 5] = s; ps2[threadIdx.x >> 5] = s2; }
  __syncthreads();
  if (threadIdx.x == 0) {
    float a = 0.f, b = 0.f;
    for (int i = 0; i < 8; ++i) { a += ps[i]; b += ps2[i]; }
    const float mu = a / (float)DMODEL;
    ps[0]  = mu;
    ps2[0] = rsqrtf(b / (float)DMODEL - mu * mu + 1e-5f);
  }
  __syncthreads();
  const float mu = ps[0], inv = ps2[0];
  _Float16* orow = out + (long)t * DMODEL;
  for (int d = threadIdx.x; d < DMODEL; d += 256)
    orow[d] = (_Float16)((row[d] - mu) * inv * sc[d] + bi[d]);
}

// ---------------------------------------------------------------------------
// Causal softmax over one score row (fp32 in -> f16 probabilities out)
// ---------------------------------------------------------------------------
__global__ __launch_bounds__(256)
void softmax_causal(const float* __restrict__ scores, _Float16* __restrict__ w)
{
  const int t = blockIdx.x, h = blockIdx.y;
  const float* srow = scores + ((long)h * T_SEQ + t) * T_SEQ;
  _Float16*    wrow = w      + ((long)h * T_SEQ + t) * T_SEQ;
  const int n = t + 1;
  __shared__ float red[8];

  float mx = -3.4e38f;
  for (int s = threadIdx.x; s < n; s += 256) mx = fmaxf(mx, srow[s]);
  #pragma unroll
  for (int o = 16; o > 0; o >>= 1) mx = fmaxf(mx, __shfl_xor(mx, o, 32));
  if ((threadIdx.x & 31) == 0) red[threadIdx.x >> 5] = mx;
  __syncthreads();
  if (threadIdx.x == 0) {
    float m = red[0];
    for (int i = 1; i < 8; ++i) m = fmaxf(m, red[i]);
    red[0] = m;
  }
  __syncthreads();
  const float M = red[0];
  __syncthreads();

  float sum = 0.f;
  for (int s = threadIdx.x; s < n; s += 256) sum += expf(srow[s] - M);
  #pragma unroll
  for (int o = 16; o > 0; o >>= 1) sum += __shfl_xor(sum, o, 32);
  if ((threadIdx.x & 31) == 0) red[threadIdx.x >> 5] = sum;
  __syncthreads();
  if (threadIdx.x == 0) {
    float a = 0.f;
    for (int i = 0; i < 8; ++i) a += red[i];
    red[0] = 1.0f / a;
  }
  __syncthreads();
  const float inv = red[0];
  for (int s = threadIdx.x; s < T_SEQ; s += 256)
    wrow[s] = (s < n) ? (_Float16)(expf(srow[s] - M) * inv) : (_Float16)0.0f;
}

// ---------------------------------------------------------------------------
// Token + position embedding
// ---------------------------------------------------------------------------
__global__ __launch_bounds__(256)
void embed_kernel(const int* __restrict__ x, const float* __restrict__ wte,
                  const float* __restrict__ wpe, float* __restrict__ h)
{
  const int t = blockIdx.x;
  const long tok = (long)x[t];
  for (int d = threadIdx.x; d < DMODEL; d += 256)
    h[(long)t * DMODEL + d] = wte[tok * DMODEL + d] + wpe[(long)t * DMODEL + d];
}

// ---------------------------------------------------------------------------
extern "C" void kernel_launch(void* const* d_in, const int* in_sizes, int n_in,
                              void* d_out, int out_size, void* d_ws, size_t ws_size,
                              hipStream_t stream)
{
  (void)in_sizes; (void)n_in; (void)out_size; (void)ws_size;
  const int*   x    = (const int*)d_in[0];
  const float* wte  = (const float*)d_in[1];
  const float* wpe  = (const float*)d_in[2];
  const float* Wq   = (const float*)d_in[3];
  const float* bq   = (const float*)d_in[4];
  const float* Wk   = (const float*)d_in[5];
  const float* bk   = (const float*)d_in[6];
  const float* Wv   = (const float*)d_in[7];
  const float* bv   = (const float*)d_in[8];
  const float* ln1s = (const float*)d_in[9];
  const float* ln1b = (const float*)d_in[10];
  const float* W1   = (const float*)d_in[11];
  const float* b1   = (const float*)d_in[12];
  const float* W2   = (const float*)d_in[13];
  const float* b2   = (const float*)d_in[14];
  const float* ln2s = (const float*)d_in[15];
  const float* ln2b = (const float*)d_in[16];
  const float* lnfs = (const float*)d_in[17];
  const float* lnfb = (const float*)d_in[18];
  const float* Wlm  = (const float*)d_in[19];
  float* logits = (float*)d_out;

  // ---- workspace carve (all regions fully rewritten every call) ----
  char* p = (char*)d_ws;
  auto carve = [&](size_t bytes) {
    char* r = p;
    p += (bytes + 255) & ~(size_t)255;
    return r;
  };
  float*    h      = (float*)   carve((size_t)T_SEQ * DMODEL * 4);
  _Float16* xn16   = (_Float16*)carve((size_t)T_SEQ * DMODEL * 2);
  _Float16* q16    = (_Float16*)carve((size_t)NHEAD * T_SEQ * HSZ * 2);
  _Float16* k16    = (_Float16*)carve((size_t)NHEAD * T_SEQ * HSZ * 2);
  _Float16* v16    = (_Float16*)carve((size_t)NHEAD * T_SEQ * HSZ * 2);
  float*    scores = (float*)   carve((size_t)NHEAD * T_SEQ * T_SEQ * 4);
  _Float16* w16    = (_Float16*)carve((size_t)NHEAD * T_SEQ * T_SEQ * 2);
  _Float16* mlp16  = (_Float16*)carve((size_t)T_SEQ * DFF * 2);

  embed_kernel<<<T_SEQ, 256, 0, stream>>>(x, wte, wpe, h);

  for (int l = 0; l < NLAYER; ++l) {
    layernorm_f16<<<T_SEQ, 256, 0, stream>>>(
        h, ln1s + (long)l * DMODEL, ln1b + (long)l * DMODEL, xn16);

    // ---- per-head QKV projections (batched over heads via grid.z) ----
    const float* Wqkv[3] = {Wq, Wk, Wv};
    const float* bqkv[3] = {bq, bk, bv};
    _Float16*    oqkv[3] = {q16, k16, v16};
    for (int j = 0; j < 3; ++j)
      gemm_wmma<EPI_F16_BIAS, B_FP32_KN>
          <<<dim3(T_SEQ / BM, 1, NHEAD), 256, 0, stream>>>(
          xn16, 0, DMODEL,
          Wqkv[j] + (long)l * NHEAD * DMODEL * HSZ, (long)DMODEL * HSZ, HSZ,
          bqkv[j] + (long)l * NHEAD * HSZ, HSZ,
          nullptr, 0, 0,
          oqkv[j], (long)T_SEQ * HSZ, HSZ,
          HSZ, DMODEL, 0);

    // ---- scores = q @ k^T (no 1/sqrt(hs) scaling, per reference) ----
    gemm_wmma<EPI_F32, B_F16_NK>
        <<<dim3(T_SEQ / BM, T_SEQ / BN, NHEAD), 256, 0, stream>>>(
        q16, (long)T_SEQ * HSZ, HSZ,
        k16, (long)T_SEQ * HSZ, HSZ,
        nullptr, 0,
        scores, (long)T_SEQ * T_SEQ, T_SEQ,
        nullptr, 0, 0,
        T_SEQ, HSZ, /*causal=*/1);

    softmax_causal<<<dim3(T_SEQ, NHEAD), 256, 0, stream>>>(scores, w16);

    // ---- attn = softmax @ v, residual-added into h at column h*64 ----
    gemm_wmma<EPI_RESID, B_F16_KN>
        <<<dim3(T_SEQ / BM, 1, NHEAD), 256, 0, stream>>>(
        w16, (long)T_SEQ * T_SEQ, T_SEQ,
        v16, (long)T_SEQ * HSZ, HSZ,
        nullptr, 0,
        h, (long)HSZ, DMODEL,
        nullptr, 0, 0,
        HSZ, T_SEQ, 0);

    layernorm_f16<<<T_SEQ, 256, 0, stream>>>(
        h, ln2s + (long)l * DMODEL, ln2b + (long)l * DMODEL, xn16);

    // ---- MLP up-projection + exact GELU ----
    gemm_wmma<EPI_F16_GELU, B_FP32_KN>
        <<<dim3(T_SEQ / BM, DFF / BN, 1), 256, 0, stream>>>(
        xn16, 0, DMODEL,
        W1 + (long)l * DMODEL * DFF, 0, DFF,
        b1 + (long)l * DFF, 0,
        nullptr, 0, 0,
        mlp16, 0, DFF,
        DFF, DMODEL, 0);

    // ---- MLP down-projection + bias + residual into h ----
    gemm_wmma<EPI_RESID, B_FP32_KN>
        <<<dim3(T_SEQ / BM, DMODEL / BN, 1), 256, 0, stream>>>(
        mlp16, 0, DFF,
        W2 + (long)l * DFF * DMODEL, 0, DMODEL,
        b2 + (long)l * DMODEL, 0,
        h, 0, DMODEL,
        nullptr, 0, 0,
        DMODEL, DFF, 0);
  }

  layernorm_f16<<<T_SEQ, 256, 0, stream>>>(h, lnfs, lnfb, xn16);

  // ---- lm_head: [1024,768] x [768,50257] -> logits (ragged N bounds) ----
  gemm_wmma<EPI_F32, B_FP32_KN, 1>
      <<<dim3(T_SEQ / BM, (VOCAB + BN - 1) / BN, 1), 256, 0, stream>>>(
      xn16, 0, DMODEL,
      Wlm, 0, VOCAB,
      nullptr, 0,
      logits, 0, VOCAB,
      nullptr, 0, 0,
      VOCAB, DMODEL, 0);
}
